// SDCrossAttention_24661702214102
// MI455X (gfx1250) — matrix operand order repaired and checked by reading, not compiled, not run on hardware
//
#include <hip/hip_runtime.h>

typedef __attribute__((ext_vector_type(16))) _Float16 v16h;
typedef __attribute__((ext_vector_type(8)))  _Float16 v8h;
typedef __attribute__((ext_vector_type(4)))  _Float16 v4h;
typedef __attribute__((ext_vector_type(8)))  float    v8f;
typedef __attribute__((ext_vector_type(4)))  float    v4f;
typedef int v4i_vs __attribute__((vector_size(4 * sizeof(int))));

constexpr int   kB   = 2;
constexpr int   kS   = 2048;
constexpr int   kD   = 640;
constexpr int   kH   = 8;
constexpr int   kDH  = 80;
constexpr int   kDHP = 96;               // dh padded to 3*32 for WMMA K-dim
constexpr float kScale = 0.11180339887498949f;  // 80^-0.5

// ---------------------------------------------------- CDNA5 async-LDS helpers

#if defined(__AMDGCN__) && __has_builtin(__builtin_amdgcn_global_load_async_to_lds_b128)
#define HAVE_ASYNC_LDS 1
#else
#define HAVE_ASYNC_LDS 0
#endif

// Copy 16 bytes global -> LDS.  Async (ASYNCcnt) when available.
__device__ __forceinline__ void copy16_g2l(_Float16* lds, const _Float16* g) {
#if HAVE_ASYNC_LDS
  __builtin_amdgcn_global_load_async_to_lds_b128(
      (__attribute__((address_space(1))) v4i_vs*)g,
      (__attribute__((address_space(3))) v4i_vs*)lds, 0, 0);
#else
  *(v8h*)lds = *(const v8h*)g;
#endif
}

__device__ __forceinline__ void async_wait0() {
#if HAVE_ASYNC_LDS
#if __has_builtin(__builtin_amdgcn_s_wait_asynccnt)
  __builtin_amdgcn_s_wait_asynccnt(0);
#else
  asm volatile("s_wait_asynccnt 0" ::: "memory");
#endif
#endif
}

// ---------------------------------------------------------------- WMMA helpers

__device__ __forceinline__ v8f wmma16(v16h a, v16h b, v8f c) {
  // D = A(16x32 f16) * B(32x16 f16) + C(16x16 f32)
  return __builtin_amdgcn_wmma_f32_16x16x32_f16(false, a, false, b, (short)0, c,
                                                false, false);
}

// A-matrix 16x32 f16 fragment.  Storage: row-major [m][k], k contiguous.
__device__ __forceinline__ v16h load_a_frag(const _Float16* base, int ld, int lane) {
  const int m = lane & 15, kh = lane >> 4;
  const _Float16* p = base + m * ld + kh * 8;
  v8h lo = *(const v8h*)(p);
  v8h hi = *(const v8h*)(p + 16);
  v16h r;
#pragma unroll
  for (int i = 0; i < 8; ++i) { r[i] = lo[i]; r[8 + i] = hi[i]; }
  return r;
}

// B-matrix 32x16 f16 fragment.  Storage: B^T row-major [n][k], k contiguous.
__device__ __forceinline__ v16h load_b_frag(const _Float16* base, int ld, int lane) {
  const int n = lane & 15, kh = lane >> 4;
  const _Float16* p = base + n * ld + kh * 8;
  v8h lo = *(const v8h*)(p);
  v8h hi = *(const v8h*)(p + 16);
  v16h r;
#pragma unroll
  for (int i = 0; i < 8; ++i) { r[i] = lo[i]; r[8 + i] = hi[i]; }
  return r;
}

// Row reductions across the 16-lane group that shares C-layout rows.
__device__ __forceinline__ float hmax16(float v) {
  v = fmaxf(v, __shfl_xor(v, 1, 32));
  v = fmaxf(v, __shfl_xor(v, 2, 32));
  v = fmaxf(v, __shfl_xor(v, 4, 32));
  v = fmaxf(v, __shfl_xor(v, 8, 32));
  return v;
}
__device__ __forceinline__ float hsum16(float v) {
  v += __shfl_xor(v, 1, 32);
  v += __shfl_xor(v, 2, 32);
  v += __shfl_xor(v, 4, 32);
  v += __shfl_xor(v, 8, 32);
  return v;
}

// ------------------------------------------------------------- projection GEMM
// C[m,n] = A[m, :640] * W[:640, n].   M=4096, N=640, K=640.
// Block tile 128x64, 8 waves 4x2, each wave a 32x32 tile (4 WMMA accums).
// mode 0/1: Q/K -> f16 [B,H,S,96]; mode 2: V -> f16 [B,H,DH,S] (transposed);
// mode 3: out-proj, A is f16, +bias, fp32 out.
__global__ __launch_bounds__(256) void proj_gemm(
    const float* __restrict__ Af, const _Float16* __restrict__ Ah,
    const float* __restrict__ W, const float* __restrict__ bias,
    _Float16* __restrict__ dstH, float* __restrict__ dstF, int mode) {
  constexpr int LDA = 40;   // 32 + 8 pad (80 B pitch)
  constexpr int LDB = 40;
  __shared__ __align__(16) _Float16 As[128 * LDA];
  __shared__ __align__(16) _Float16 Bs[64 * LDB];

  const int tid = threadIdx.x, lane = tid & 31, wave = tid >> 5;
  const int wm = wave >> 1, wn = wave & 1;
  const int m0 = blockIdx.y * 128, n0 = blockIdx.x * 64;

  v8f acc[2][2] = {};

  for (int kt = 0; kt < kD / 32; ++kt) {
    const int k0 = kt * 32;
    __syncthreads();
    // ---- stage A tile 128x32: issue ALL loads, then store (keeps MLP)
    if (mode < 3) {
      v4f t[4];
#pragma unroll
      for (int u = 0; u < 4; ++u) {
        int i = tid + u * 256, row = i >> 3, kc = (i & 7) << 2;
        t[u] = *(const v4f*)(Af + (size_t)(m0 + row) * kD + k0 + kc);
      }
#pragma unroll
      for (int u = 0; u < 4; ++u) {
        int i = tid + u * 256, row = i >> 3, kc = (i & 7) << 2;
        v4h h; h[0] = (_Float16)t[u][0]; h[1] = (_Float16)t[u][1];
        h[2] = (_Float16)t[u][2]; h[3] = (_Float16)t[u][3];
        *(v4h*)&As[row * LDA + kc] = h;
      }
    } else {
      v4h t[4];
#pragma unroll
      for (int u = 0; u < 4; ++u) {
        int i = tid + u * 256, row = i >> 3, kc = (i & 7) << 2;
        t[u] = *(const v4h*)(Ah + (size_t)(m0 + row) * kD + k0 + kc);
      }
#pragma unroll
      for (int u = 0; u < 4; ++u) {
        int i = tid + u * 256, row = i >> 3, kc = (i & 7) << 2;
        *(v4h*)&As[row * LDA + kc] = t[u];
      }
    }
    // ---- stage W tile as W^T [n][k] (transposed scatter into LDS)
    {
      v4f t[2];
#pragma unroll
      for (int u = 0; u < 2; ++u) {
        int i = tid + u * 256, kk = i >> 4, nb = (i & 15) << 2;
        t[u] = *(const v4f*)(W + (size_t)(k0 + kk) * kD + n0 + nb);
      }
#pragma unroll
      for (int u = 0; u < 2; ++u) {
        int i = tid + u * 256, kk = i >> 4, nb = (i & 15) << 2;
        Bs[(nb + 0) * LDB + kk] = (_Float16)t[u][0];
        Bs[(nb + 1) * LDB + kk] = (_Float16)t[u][1];
        Bs[(nb + 2) * LDB + kk] = (_Float16)t[u][2];
        Bs[(nb + 3) * LDB + kk] = (_Float16)t[u][3];
      }
    }
    if (kt + 1 < kD / 32 && mode < 3)
      __builtin_prefetch(Af + (size_t)(m0 + (tid >> 1)) * kD + k0 + 32, 0, 1);
    __syncthreads();
    // ---- 4 WMMAs for this K-slab
    v16h a0 = load_a_frag(&As[(wm * 32) * LDA], LDA, lane);
    v16h a1 = load_a_frag(&As[(wm * 32 + 16) * LDA], LDA, lane);
    v16h b0 = load_b_frag(&Bs[(wn * 32) * LDB], LDB, lane);
    v16h b1 = load_b_frag(&Bs[(wn * 32 + 16) * LDB], LDB, lane);
    acc[0][0] = wmma16(a0, b0, acc[0][0]);
    acc[0][1] = wmma16(a0, b1, acc[0][1]);
    acc[1][0] = wmma16(a1, b0, acc[1][0]);
    acc[1][1] = wmma16(a1, b1, acc[1][1]);
  }

  // ---- epilogue: C layout lane = col (lane&15), rows (lane>>4)*8 + r
  const int nl = lane & 15, mh = lane >> 4;
#pragma unroll
  for (int i = 0; i < 2; ++i)
#pragma unroll
    for (int j = 0; j < 2; ++j)
#pragma unroll
      for (int r = 0; r < 8; ++r) {
        int m = m0 + wm * 32 + i * 16 + mh * 8 + r;
        int n = n0 + wn * 32 + j * 16 + nl;
        float v = acc[i][j][r];
        if (mode == 3) {
          dstF[(size_t)m * kD + n] = v + bias[n];
        } else {
          int bb = m >> 11, s = m & (kS - 1);
          int h = n / kDH, dh = n % kDH;
          if (mode == 2)
            dstH[((size_t)(bb * kH + h) * kDH + dh) * kS + s] = (_Float16)v;
          else
            dstH[((size_t)(bb * kH + h) * kS + s) * kDHP + dh] = (_Float16)v;
        }
      }
}

// ------------------------------------------------------------- flash attention
// Block: 128 query rows, 8 waves x 16 rows each.  Inner step: 32 keys.
// K/V tiles double-buffered in LDS; staged with async global->LDS copies
// (ASYNCcnt) issued one step ahead so DMA overlaps WMMA+softmax compute.
__global__ __launch_bounds__(256) void attn_kernel(
    const _Float16* __restrict__ Qb, const _Float16* __restrict__ Kb,
    const _Float16* __restrict__ Vt, _Float16* __restrict__ Ob) {
  constexpr int LDK = 104;  // 96 + 8 pad  (208 B pitch)
  constexpr int LDV = 40;   // 32 keys + 8 pad
  constexpr int LDP = 40;
  __shared__ __align__(16) _Float16 Ks[2][32 * LDK];
  __shared__ __align__(16) _Float16 Vs[2][kDH * LDV];
  __shared__ __align__(16) _Float16 Ps[8 * 16 * LDP];

  const int tid = threadIdx.x, lane = tid & 31, wave = tid >> 5;
  const int nl = lane & 15, kh = lane >> 4;
  const int bh = blockIdx.y;            // b*H + h
  const int q0 = blockIdx.x * 128;

  const _Float16* Kbase = Kb + (size_t)bh * kS * kDHP;
  const _Float16* Vbase = Vt + (size_t)bh * kDH * kS;

  // Stage one 32-key K/V tile into LDS buffer `buf` (16B async chunks).
  auto stage = [&](int buf, int s0) {
    // K tile: 32 keys x 96 dh = 32 rows x 12 b128 chunks
    for (int i = tid; i < 384; i += 256) {
      int key = i / 12, c = i % 12;
      copy16_g2l(&Ks[buf][key * LDK + c * 8],
                 Kbase + (size_t)(s0 + key) * kDHP + c * 8);
    }
    // V tile: 80 dh x 32 keys = 80 rows x 4 b128 chunks
    for (int i = tid; i < 320; i += 256) {
      int dh = i >> 2, c = i & 3;
      copy16_g2l(&Vs[buf][dh * LDV + c * 8],
                 Vbase + (size_t)dh * kS + s0 + c * 8);
    }
  };

  // Q fragments for this wave's 16 rows: resident for the whole kernel.
  const _Float16* Qrow = Qb + ((size_t)bh * kS + q0 + wave * 16) * kDHP;
  v16h qf[3];
#pragma unroll
  for (int c = 0; c < 3; ++c) qf[c] = load_a_frag(Qrow + c * 32, kDHP, lane);

  v8f o[5] = {};
  float mr[8], lr[8];
#pragma unroll
  for (int r = 0; r < 8; ++r) { mr[r] = -1e30f; lr[r] = 0.0f; }

  stage(0, 0);
  async_wait0();
  __syncthreads();

  for (int kt = 0; kt < kS / 32; ++kt) {
    const int buf = kt & 1;
    if (kt + 1 < kS / 32) stage(buf ^ 1, (kt + 1) * 32);  // prefetch next tile

    // ---- scores: S(16q x 32keys) as two 16x16 C-frags, K-dim = 96 (3 slabs)
    v8f sf0 = {}, sf1 = {};
#pragma unroll
    for (int c = 0; c < 3; ++c) {
      v16h b0 = load_b_frag(&Ks[buf][0 * LDK + c * 32], LDK, lane);
      v16h b1 = load_b_frag(&Ks[buf][16 * LDK + c * 32], LDK, lane);
      sf0 = wmma16(qf[c], b0, sf0);
      sf1 = wmma16(qf[c], b1, sf1);
    }

    // ---- online softmax (f32 stats; rows shared by 16-lane halves)
    float corr[8];
#pragma unroll
    for (int r = 0; r < 8; ++r) {
      float v0 = sf0[r] * kScale, v1 = sf1[r] * kScale;
      float mx = hmax16(fmaxf(v0, v1));
      float mn = fmaxf(mr[r], mx);
      corr[r] = __expf(mr[r] - mn);
      mr[r] = mn;
      float p0 = __expf(v0 - mn), p1 = __expf(v1 - mn);
      sf0[r] = p0; sf1[r] = p1;
      lr[r] = lr[r] * corr[r] + hsum16(p0 + p1);
    }
#pragma unroll
    for (int d = 0; d < 5; ++d)
#pragma unroll
      for (int r = 0; r < 8; ++r) o[d][r] *= corr[r];

    // ---- C-layout -> A-fragment conversion through per-wave LDS tile.
    // Same-wave DS ops are in-order: write-then-read needs no barrier.
    _Float16* Pw = &Ps[wave * 16 * LDP];
#pragma unroll
    for (int r = 0; r < 8; ++r) {
      Pw[(kh * 8 + r) * LDP + nl]      = (_Float16)sf0[r];
      Pw[(kh * 8 + r) * LDP + 16 + nl] = (_Float16)sf1[r];
    }
    v16h pa = load_a_frag(Pw, LDP, lane);

    // ---- O += P(16x32) @ V(32keys x 80dh), 5 dh tiles of 16
#pragma unroll
    for (int d = 0; d < 5; ++d) {
      v16h vb = load_b_frag(&Vs[buf][(d * 16) * LDV], LDV, lane);
      o[d] = wmma16(pa, vb, o[d]);
    }

    async_wait0();     // next tile's DMA complete (this wave)
    __syncthreads();   // ... and visible to / done by all waves
  }

  // ---- normalize and write f16 [B,S,H*DH] (== flattened [4096,640])
  const int bb = bh >> 3, hh = bh & 7;
#pragma unroll
  for (int r = 0; r < 8; ++r) {
    float inv = 1.0f / lr[r];
    int sg = q0 + wave * 16 + kh * 8 + r;
    _Float16* dst = Ob + ((size_t)bb * kS + sg) * kD + hh * kDH;
#pragma unroll
    for (int d = 0; d < 5; ++d) dst[d * 16 + nl] = (_Float16)(o[d][r] * inv);
  }
}

// ----------------------------------------------------------------- launch glue

extern "C" void kernel_launch(void* const* d_in, const int* in_sizes, int n_in,
                              void* d_out, int out_size, void* d_ws, size_t ws_size,
                              hipStream_t stream) {
  const float* x  = (const float*)d_in[0];
  const float* eh = (const float*)d_in[1];
  const float* Wq = (const float*)d_in[2];
  const float* Wk = (const float*)d_in[3];
  const float* Wv = (const float*)d_in[4];
  const float* Wo = (const float*)d_in[5];
  const float* bo = (const float*)d_in[6];
  float* out = (float*)d_out;

  // workspace layout (f16): Qb/Kb [B,H,S,96], Vt [B,H,DH,S], Ob [B,S,640]
  const size_t qkElems = (size_t)kB * kH * kS * kDHP;     // 3,145,728
  const size_t vElems  = (size_t)kB * kH * kDH * kS;      // 2,621,440
  const size_t oElems  = (size_t)kB * kS * kD;            // 2,621,440
  char* ws = (char*)d_ws;
  _Float16* Qb = (_Float16*)(ws);
  _Float16* Kb = (_Float16*)(ws + qkElems * 2);
  _Float16* Vt = (_Float16*)(ws + qkElems * 4);
  _Float16* Ob = (_Float16*)(ws + qkElems * 4 + vElems * 2);
  if (ws_size < qkElems * 4 + vElems * 2 + oElems * 2) return;

  // zero the dh-pad region of Q/K (GEMM only writes dh<80)
  (void)hipMemsetAsync(Qb, 0, qkElems * sizeof(_Float16), stream);
  (void)hipMemsetAsync(Kb, 0, qkElems * sizeof(_Float16), stream);

  dim3 blk(256);
  dim3 gGemm(kD / 64, (kB * kS) / 128);   // 10 x 32
  proj_gemm<<<gGemm, blk, 0, stream>>>(x,  nullptr, Wq, nullptr, Qb, nullptr, 0);
  proj_gemm<<<gGemm, blk, 0, stream>>>(eh, nullptr, Wk, nullptr, Kb, nullptr, 1);
  proj_gemm<<<gGemm, blk, 0, stream>>>(eh, nullptr, Wv, nullptr, Vt, nullptr, 2);

  dim3 gAttn(kS / 128, kB * kH);          // 16 x 16
  attn_kernel<<<gAttn, blk, 0, stream>>>(Qb, Kb, Vt, Ob);

  proj_gemm<<<gGemm, blk, 0, stream>>>(nullptr, Ob, Wo, bo, nullptr, out, 3);
}
